// DebiasedPosLossV2_2662879724031
// MI455X (gfx1250) — matrix operand order repaired and checked by reading, not compiled, not run on hardware
//
#include <hip/hip_runtime.h>

// Problem constants (from reference)
#define BBATCH 4096
#define DDIM   128
#define NROWS  8192                       // 2*B
#define CSPLIT 8                          // column splits for parallelism
// (1/TEMPERATURE) * log2(e) with T = 0.5
#define KSCALE 2.8853900817779268f
#define LN2    0.6931471805599453f

typedef __attribute__((ext_vector_type(16))) __bf16 v16bf;
typedef __attribute__((ext_vector_type(8)))  __bf16 v8bf;
typedef __attribute__((ext_vector_type(8)))  float  v8f;

union BfBits { __bf16 b; unsigned short u; };

// ---------------------------------------------------------------------------
// Kernel 1: fp32 -> (hi, lo) bf16 split of out = concat(out_1, out_2)
// ---------------------------------------------------------------------------
__global__ __launch_bounds__(256) void pack_kernel(const float* __restrict__ o1,
                                                   const float* __restrict__ o2,
                                                   unsigned short* __restrict__ hi,
                                                   unsigned short* __restrict__ lo) {
  int idx = blockIdx.x * 256 + threadIdx.x;          // grid covers exactly 2B*D
  const float* src = (idx < BBATCH * DDIM) ? o1 : (o2 - BBATCH * DDIM);
  float x = src[idx];
  BfBits h; h.b = (__bf16)x;
  float hf = (float)h.b;
  BfBits l; l.b = (__bf16)(x - hf);
  hi[idx] = h.u;
  lo[idx] = l.u;
}

// ---------------------------------------------------------------------------
// helper: load one 32-K B chunk (two 8-bf16 runs at p and p+16 per lane) as v16bf
// ---------------------------------------------------------------------------
__device__ __forceinline__ v16bf loadB(const __bf16* p) {
  v8bf a = *reinterpret_cast<const v8bf*>(p);
  v8bf b = *reinterpret_cast<const v8bf*>(p + 16);
  return __builtin_shufflevector(a, b, 0,1,2,3,4,5,6,7,8,9,10,11,12,13,14,15);
}

// ---------------------------------------------------------------------------
// Kernel 2: fused  exp((X X^T)/T)  row-sum (full + masked Ng), WMMA bf16 split
//   grid = (NROWS/128, CSPLIT), block = 256 (8 waves; 1 wave = 16 rows)
//   Each wave iterates its column slice 32 columns (2 independent 16x16
//   accumulator chains) at a time for matrix-op ILP.
// ---------------------------------------------------------------------------
__global__ __launch_bounds__(256) void sim_rowsum_kernel(const unsigned short* __restrict__ Xhi_u,
                                                         const unsigned short* __restrict__ Xlo_u,
                                                         const int* __restrict__ tgt,
                                                         float* __restrict__ fullPart,
                                                         float* __restrict__ ngPart) {
  const __bf16* Xhi = reinterpret_cast<const __bf16*>(Xhi_u);
  const __bf16* Xlo = reinterpret_cast<const __bf16*>(Xlo_u);

  const int lane = threadIdx.x & 31;
  const int wave = threadIdx.x >> 5;
  const int half = lane >> 4;          // which 16-lane half of the wave32
  const int l16  = lane & 15;
  const int rowBase = blockIdx.x * 128 + wave * 16;

  // ---- A operand: 16 rows x K=128 (hi & lo), resident in registers -------
  // 16-bit A layout: lane (l16) = row M; elements 0..7 -> K = half*8 + e,
  //                  elements 8..15 -> K = half*8 + 16 + (e-8).  (per 32-K chunk)
  v16bf Ahi[4], Alo[4];
  {
    const __bf16* ra = Xhi + (size_t)(rowBase + l16) * DDIM;
    const __bf16* rb = Xlo + (size_t)(rowBase + l16) * DDIM;
#pragma unroll
    for (int c = 0; c < 4; ++c) {
      int k0 = 32 * c + half * 8;
      v8bf h0 = *reinterpret_cast<const v8bf*>(ra + k0);
      v8bf h1 = *reinterpret_cast<const v8bf*>(ra + k0 + 16);
      Ahi[c] = __builtin_shufflevector(h0, h1, 0,1,2,3,4,5,6,7,8,9,10,11,12,13,14,15);
      v8bf l0 = *reinterpret_cast<const v8bf*>(rb + k0);
      v8bf l1 = *reinterpret_cast<const v8bf*>(rb + k0 + 16);
      Alo[c] = __builtin_shufflevector(l0, l1, 0,1,2,3,4,5,6,7,8,9,10,11,12,13,14,15);
    }
  }

  // Per-lane row metadata for C layout: VGPR v holds row M = half*8 + v.
  int rmod[8]; int tmv[8];
#pragma unroll
  for (int v = 0; v < 8; ++v) {
    int m = rowBase + half * 8 + v;
    rmod[v] = m & (BBATCH - 1);
    tmv[v]  = tgt[m & (BBATCH - 1)];
  }

  float accF[8], accN[8];
#pragma unroll
  for (int v = 0; v < 8; ++v) { accF[v] = 0.f; accN[v] = 0.f; }

  // Column slice for this blockIdx.y: NROWS/CSPLIT = 1024 columns,
  // processed 32 at a time (two 16-wide tiles per iteration).
  const int colBeg = blockIdx.y * (NROWS / CSPLIT);
  const int nIter  = (NROWS / CSPLIT) / 32;

  int n0 = colBeg + l16;               // tile-0 column for this lane
  const __bf16* cbh0 = Xhi + (size_t)n0 * DDIM + half * 8;
  const __bf16* cbl0 = Xlo + (size_t)n0 * DDIM + half * 8;

  for (int it = 0; it < nIter; ++it) {
    // B layout: lane = column N (l16); K = half*8 + e (e<8), half*8 + 16 + (e-8) per chunk.
    const __bf16* cbh1 = cbh0 + (size_t)16 * DDIM;   // tile-1 = tile-0 + 16 cols
    const __bf16* cbl1 = cbl0 + (size_t)16 * DDIM;

    int nmod0 = n0 & (BBATCH - 1);
    int nmod1 = (n0 + 16) & (BBATCH - 1);
    int tn0 = tgt[nmod0];
    int tn1 = tgt[nmod1];

    if (it + 1 < nIter) {              // prefetch next iteration's B rows to L0
      __builtin_prefetch(cbh0 + (size_t)32 * DDIM, 0, 3);
      __builtin_prefetch(cbl0 + (size_t)32 * DDIM, 0, 3);
      __builtin_prefetch(cbh1 + (size_t)32 * DDIM, 0, 3);
      __builtin_prefetch(cbl1 + (size_t)32 * DDIM, 0, 3);
    }

    v8f acc0 = {};
    v8f acc1 = {};
#pragma unroll
    for (int c = 0; c < 4; ++c) {
      v16bf B0h = loadB(cbh0 + 32 * c);
      v16bf B0l = loadB(cbl0 + 32 * c);
      v16bf B1h = loadB(cbh1 + 32 * c);
      v16bf B1l = loadB(cbl1 + 32 * c);
      // Two independent accumulator chains; hi*hi + hi*lo + lo*hi per chain.
      acc0 = __builtin_amdgcn_wmma_f32_16x16x32_bf16(false, Ahi[c], false, B0h, (short)0, acc0, false, false);
      acc1 = __builtin_amdgcn_wmma_f32_16x16x32_bf16(false, Ahi[c], false, B1h, (short)0, acc1, false, false);
      acc0 = __builtin_amdgcn_wmma_f32_16x16x32_bf16(false, Ahi[c], false, B0l, (short)0, acc0, false, false);
      acc1 = __builtin_amdgcn_wmma_f32_16x16x32_bf16(false, Ahi[c], false, B1l, (short)0, acc1, false, false);
      acc0 = __builtin_amdgcn_wmma_f32_16x16x32_bf16(false, Alo[c], false, B0h, (short)0, acc0, false, false);
      acc1 = __builtin_amdgcn_wmma_f32_16x16x32_bf16(false, Alo[c], false, B1h, (short)0, acc1, false, false);
    }

#pragma unroll
    for (int v = 0; v < 8; ++v) {
      float s0 = __builtin_amdgcn_exp2f(acc0[v] * KSCALE);   // exp(dot/T)
      float s1 = __builtin_amdgcn_exp2f(acc1[v] * KSCALE);
      accF[v] += s0 + s1;
      bool k0 = (rmod[v] != nmod0) && (tmv[v] != tn0);
      bool k1 = (rmod[v] != nmod1) && (tmv[v] != tn1);
      accN[v] += (k0 ? s0 : 0.f) + (k1 ? s1 : 0.f);
    }

    n0   += 32;
    cbh0 += (size_t)32 * DDIM;
    cbl0 += (size_t)32 * DDIM;
  }

  // Reduce across the 16 lanes that share a row, then deterministic store.
#pragma unroll
  for (int v = 0; v < 8; ++v) {
    float f = accF[v], g = accN[v];
#pragma unroll
    for (int msk = 8; msk >= 1; msk >>= 1) {
      f += __shfl_xor(f, msk, 32);
      g += __shfl_xor(g, msk, 32);
    }
    if (l16 == 0) {
      int m = rowBase + half * 8 + v;
      fullPart[blockIdx.y * NROWS + m] = f;
      ngPart[blockIdx.y * NROWS + m]   = g;
    }
  }
}

// ---------------------------------------------------------------------------
// Kernel 3: fold partials per row, compute -log(o1/o2), mean -> scalar
// ---------------------------------------------------------------------------
__global__ __launch_bounds__(1024) void loss_kernel(const float* __restrict__ fullPart,
                                                    const float* __restrict__ ngPart,
                                                    float* __restrict__ out) {
  __shared__ float red[1024];
  float s = 0.f;
  for (int i = threadIdx.x; i < NROWS; i += 1024) {
    float F = 0.f, G = 0.f;
#pragma unroll
    for (int p = 0; p < CSPLIT; ++p) {
      F += fullPart[p * NROWS + i];
      G += ngPart[p * NROWS + i];
    }
    float o1 = F - 0.9f * G;         // (1 - TAU_PLUS)
    float o2 = F + 818.1f * G;       // n*TAU_PLUS - (1 - TAU_PLUS), n = 8190
    // -log(o1/o2) = log(o2) - log(o1);  v_log_f32 is log2 -> scale by ln2
    s += (__builtin_amdgcn_logf(o2) - __builtin_amdgcn_logf(o1)) * LN2;
  }
  red[threadIdx.x] = s;
  __syncthreads();
  for (int st = 512; st > 0; st >>= 1) {
    if (threadIdx.x < st) red[threadIdx.x] += red[threadIdx.x + st];
    __syncthreads();
  }
  if (threadIdx.x == 0) out[0] = red[0] * (1.0f / (float)NROWS);
}

// ---------------------------------------------------------------------------
extern "C" void kernel_launch(void* const* d_in, const int* in_sizes, int n_in,
                              void* d_out, int out_size, void* d_ws, size_t ws_size,
                              hipStream_t stream) {
  const float* out1 = (const float*)d_in[0];
  const float* out2 = (const float*)d_in[1];
  // d_in[2] (out_m) is unused by the loss math
  const int* tgt = (const int*)d_in[3];

  char* ws = (char*)d_ws;
  const size_t bfBytes = (size_t)NROWS * DDIM * sizeof(unsigned short);   // 2 MiB each
  unsigned short* hi = (unsigned short*)ws;
  unsigned short* lo = (unsigned short*)(ws + bfBytes);
  float* fullPart = (float*)(ws + 2 * bfBytes);                           // CSPLIT*NROWS
  float* ngPart   = (float*)(ws + 2 * bfBytes + (size_t)CSPLIT * NROWS * sizeof(float));

  pack_kernel<<<(NROWS * DDIM) / 256, 256, 0, stream>>>(out1, out2, hi, lo);
  sim_rowsum_kernel<<<dim3(NROWS / 128, CSPLIT), 256, 0, stream>>>(hi, lo, tgt, fullPart, ngPart);
  loss_kernel<<<1, 1024, 0, stream>>>(fullPart, ngPart, (float*)d_out);
}
